// WindowMultiHeadAttentionNoPos_8821862826385
// MI455X (gfx1250) — matrix operand order repaired and checked by reading, not compiled, not run on hardware
//
#include <hip/hip_runtime.h>
#include <hip/hip_bf16.h>
#include <math.h>

typedef __attribute__((ext_vector_type(16))) _Float16 v16h;
typedef __attribute__((ext_vector_type(8)))  _Float16 v8h;
typedef __attribute__((ext_vector_type(8)))  float    v8f;

#define BW 1024
#define LW 144
#define CC 512
#define HH 16
#define DH 32
#define NW 64
#define MM (BW * LW)        // 147456
#define NQKV (3 * CC)       // 1536

// ---------------------------------------------------------------------------
// WMMA fragment helpers (layouts per CDNA5 ISA 7.12.2, wave32)
// ---------------------------------------------------------------------------

// A fragment (16x32 f16, M x K): lane&15 = row, (lane>>4) selects K groups
__device__ inline v16h load_a_f16(const _Float16* src, int ld, int lane) {
    int row = lane & 15;
    int kb  = (lane >> 4) * 8;
    v8h lo = *(const v8h*)(src + row * ld + kb);
    v8h hi = *(const v8h*)(src + row * ld + 16 + kb);
    v16h a;
#pragma unroll
    for (int i = 0; i < 8; ++i) { a[i] = lo[i]; a[8 + i] = hi[i]; }
    return a;
}

// B fragment (32x16 f16, K x N) from an N-major buffer (row = n, contiguous K)
__device__ inline v16h load_b_f16(const _Float16* srcNmajor, int ld, int lane) {
    int n  = lane & 15;
    int kb = (lane >> 4) * 8;
    v8h lo = *(const v8h*)(srcNmajor + n * ld + kb);
    v8h hi = *(const v8h*)(srcNmajor + n * ld + 16 + kb);
    v16h b;
#pragma unroll
    for (int i = 0; i < 8; ++i) { b[i] = lo[i]; b[8 + i] = hi[i]; }
    return b;
}

__device__ inline v8f wmma_f16(v16h a, v16h b, v8f c) {
    return __builtin_amdgcn_wmma_f32_16x16x32_f16(false, a, false, b,
                                                  (short)0, c, false, false);
}

// stage 16 f32 -> 16 f16 into LDS (one half-row per thread)
__device__ inline void stage_cvt16(const float* src, _Float16* dst) {
    const float4* p = (const float4*)src;
    float4 f0 = p[0], f1 = p[1], f2 = p[2], f3 = p[3];
    dst[0]  = (_Float16)f0.x; dst[1]  = (_Float16)f0.y;
    dst[2]  = (_Float16)f0.z; dst[3]  = (_Float16)f0.w;
    dst[4]  = (_Float16)f1.x; dst[5]  = (_Float16)f1.y;
    dst[6]  = (_Float16)f1.z; dst[7]  = (_Float16)f1.w;
    dst[8]  = (_Float16)f2.x; dst[9]  = (_Float16)f2.y;
    dst[10] = (_Float16)f2.z; dst[11] = (_Float16)f2.w;
    dst[12] = (_Float16)f3.x; dst[13] = (_Float16)f3.y;
    dst[14] = (_Float16)f3.z; dst[15] = (_Float16)f3.w;
}

// ---------------------------------------------------------------------------
// Kernel 0: weight f32->f16 pack + logit scale
// ---------------------------------------------------------------------------
__global__ void prep_kernel(const float* __restrict__ Wq,
                            const float* __restrict__ Wk,
                            const float* __restrict__ Wv,
                            const float* __restrict__ Wp,
                            const float* __restrict__ logit_scale,
                            _Float16* __restrict__ Wcat,   // [1536][512]
                            _Float16* __restrict__ Wph,    // [512][512]
                            float* __restrict__ scales) {  // [16]
    int i = blockIdx.x * blockDim.x + threadIdx.x;   // 0 .. 262143
    Wcat[i]          = (_Float16)Wq[i];
    Wcat[262144 + i] = (_Float16)Wk[i];
    Wcat[524288 + i] = (_Float16)Wv[i];
    Wph[i]           = (_Float16)Wp[i];
    if (i < HH) {
        const float ln100 = 4.605170185988091f;
        scales[i] = __expf(fminf(logit_scale[i], ln100));
    }
}

// ---------------------------------------------------------------------------
// Kernel 1: fused QKV GEMM with L2-normalize of q,k and head scaling.
// block = 256 (8 waves), tile = 128(M) x 64(N), wave = 32x32 sub-tile
// grid = (M/128, 1536/64).  Explicit double-buffered f32->f16 staging of x.
// ---------------------------------------------------------------------------
__global__ void qkv_gemm_kernel(const float* __restrict__ x,      // [M][512]
                                const _Float16* __restrict__ Wcat,// [1536][512]
                                const float* __restrict__ bq,
                                const float* __restrict__ bk,
                                const float* __restrict__ bv,
                                const float* __restrict__ scales, // [16]
                                _Float16* __restrict__ qkvh) {    // [M][1536]
    __shared__ __align__(32) _Float16 Abuf0[128 * 32];     // 8 KB
    __shared__ __align__(32) _Float16 Abuf1[128 * 32];     // 8 KB
    __shared__ __align__(16) float    Dtile[128 * 64];     // 32 KB
    __shared__ float invn[256];

    const int m0   = blockIdx.x * 128;
    const int nG   = blockIdx.y * 64;
    const int tid  = threadIdx.x;
    const int wave = tid >> 5;
    const int lane = tid & 31;
    const int wm   = wave >> 1;              // 0..3  (32-row group)
    const int wn   = wave & 1;               // 0..1  (32-col group)

    const int arow = tid >> 1;               // 0..127
    const int acol = (tid & 1) * 16;         // 0 / 16
    const float*    xsrc = x + (size_t)(m0 + arow) * CC + acol;
    _Float16* const d0   = Abuf0 + arow * 32 + acol;
    _Float16* const d1   = Abuf1 + arow * 32 + acol;

    const _Float16* wb0 = Wcat + (size_t)(nG + wn * 32) * CC;
    const _Float16* wb1 = Wcat + (size_t)(nG + wn * 32 + 16) * CC;
    const _Float16* a0p = Abuf0 + (wm * 32) * 32;
    const _Float16* a1p = Abuf0 + (wm * 32 + 16) * 32;
    const _Float16* a0q = Abuf1 + (wm * 32) * 32;
    const _Float16* a1q = Abuf1 + (wm * 32 + 16) * 32;

    v8f acc[2][2] = {};

    stage_cvt16(xsrc, d0);                   // k = 0 into buffer 0
    __syncthreads();

#pragma unroll 1
    for (int k0 = 0; k0 < CC; k0 += 64) {
        // ---- stage k0+32 into buffer 1 (always valid: k0+32 <= 480) ----
        stage_cvt16(xsrc + k0 + 32, d1);
        __builtin_prefetch(xsrc + k0 + 64, 0, 3);
        {   // compute from buffer 0 at k0
            v16h a0 = load_a_f16(a0p, 32, lane);
            v16h a1 = load_a_f16(a1p, 32, lane);
            v16h b0 = load_b_f16(wb0 + k0, CC, lane);
            v16h b1 = load_b_f16(wb1 + k0, CC, lane);
            acc[0][0] = wmma_f16(a0, b0, acc[0][0]);
            acc[0][1] = wmma_f16(a0, b1, acc[0][1]);
            acc[1][0] = wmma_f16(a1, b0, acc[1][0]);
            acc[1][1] = wmma_f16(a1, b1, acc[1][1]);
        }
        __syncthreads();
        // ---- stage k0+64 into buffer 0 (skip on final pass) ----
        if (k0 + 64 < CC) stage_cvt16(xsrc + k0 + 64, d0);
        {   // compute from buffer 1 at k0+32
            v16h a0 = load_a_f16(a0q, 32, lane);
            v16h a1 = load_a_f16(a1q, 32, lane);
            v16h b0 = load_b_f16(wb0 + k0 + 32, CC, lane);
            v16h b1 = load_b_f16(wb1 + k0 + 32, CC, lane);
            acc[0][0] = wmma_f16(a0, b0, acc[0][0]);
            acc[0][1] = wmma_f16(a0, b1, acc[0][1]);
            acc[1][0] = wmma_f16(a1, b0, acc[1][0]);
            acc[1][1] = wmma_f16(a1, b1, acc[1][1]);
        }
        __syncthreads();
    }

    // ---- post: bias, then per-token L2 norm over each 32-wide head half ----
    const int proj = nG >> 9;                 // 0=q 1=k 2=v
    const int nc0  = nG & 511;
    const float* bias = (proj == 0) ? bq : (proj == 1) ? bk : bv;

#pragma unroll
    for (int ms = 0; ms < 2; ++ms)
#pragma unroll
        for (int ns = 0; ns < 2; ++ns) {
            int col = wn * 32 + ns * 16 + (lane & 15);
            float bval = bias[nc0 + col];
#pragma unroll
            for (int r = 0; r < 8; ++r) {
                int row = wm * 32 + ms * 16 + ((lane >> 4) * 8) + r;
                Dtile[row * 64 + col] = acc[ms][ns][r] + bval;
            }
        }
    __syncthreads();

    {
        int r  = tid >> 1;                    // 0..127
        int hh = tid & 1;                     // head half within 64-col tile
        const float* rp = Dtile + r * 64 + hh * 32;
        float s = 0.f;
#pragma unroll
        for (int i = 0; i < 32; ++i) s += rp[i] * rp[i];
        float inv = 1.f;
        if (proj != 2) {
            inv = 1.f / fmaxf(sqrtf(s), 1e-12f);
            if (proj == 0) inv *= scales[(nc0 >> 5) + hh];
        }
        invn[tid] = inv;
    }
    __syncthreads();

    {
        int r  = tid >> 1;
        int hh = tid & 1;
        float inv = invn[tid];
        const float* dp = Dtile + r * 64 + hh * 32;
        _Float16* op = qkvh + (size_t)(m0 + r) * NQKV + nG + hh * 32;
#pragma unroll
        for (int i = 0; i < 32; ++i) op[i] = (_Float16)(dp[i] * inv);
    }
}

// ---------------------------------------------------------------------------
// Kernel 2: fused attention per (window b, head h).  block = 256 (8 waves).
// ---------------------------------------------------------------------------
__global__ void attn_kernel(const _Float16* __restrict__ qkvh, // [M][1536]
                            const float* __restrict__ mask,    // [64][144][144]
                            _Float16* __restrict__ Oh) {       // [M][512]
    __shared__ __align__(32) _Float16 qs[LW * 32];        //  9 KB
    __shared__ __align__(32) _Float16 ks[LW * 32];        //  9 KB
    __shared__ __align__(32) _Float16 vts[32 * 160];      // 10 KB (V^T, K pad)
    __shared__ __align__(16) float    S[LW * 160];        // 90 KB

    const int b = blockIdx.x;
    const int h = blockIdx.y;
    const int w = b & (NW - 1);
    const int wave = threadIdx.x >> 5;
    const int lane = threadIdx.x & 31;

    const _Float16* qg = qkvh + (size_t)b * LW * NQKV + h * DH;
    const _Float16* kg = qg + CC;
    const _Float16* vg = qg + 2 * CC;

    for (int c = threadIdx.x; c < (LW * 32) / 8; c += 256) {
        int row = c >> 2;
        int col = (c & 3) * 8;
        *(v8h*)(qs + row * 32 + col) =
            *(const v8h*)(qg + (size_t)row * NQKV + col);
        *(v8h*)(ks + row * 32 + col) =
            *(const v8h*)(kg + (size_t)row * NQKV + col);
    }
    for (int i = threadIdx.x; i < LW * 32; i += 256) {
        int row = i >> 5;
        int d   = i & 31;
        vts[d * 160 + row] = vg[(size_t)row * NQKV + d];
    }
    for (int i = threadIdx.x; i < 32 * 16; i += 256) {
        int d   = i >> 4;
        int col = LW + (i & 15);
        vts[d * 160 + col] = (_Float16)0.f;
    }
    __syncthreads();

    // ---- S = qn @ kn^T + mask : 81 WMMA tiles, K = 32 ----
    const float* maskp = mask + (size_t)w * LW * LW;
    for (int t = wave; t < 81; t += 8) {
        int mt = t / 9, nt = t % 9;
        v16h a  = load_a_f16(qs + mt * 16 * 32, 32, lane);
        v16h bf = load_b_f16(ks + (nt * 16) * 32, 32, lane);
        v8f c = {};
        c = wmma_f16(a, bf, c);
        int ncol = nt * 16 + (lane & 15);
#pragma unroll
        for (int r = 0; r < 8; ++r) {
            int mrow = mt * 16 + ((lane >> 4) * 8) + r;
            S[mrow * 160 + ncol] = c[r] + maskp[mrow * LW + ncol];
        }
    }
    __syncthreads();

    // ---- wave-parallel row softmax (shfl_xor reductions, wave32) ----
    for (int row = wave; row < LW; row += 8) {
        float* rp = S + row * 160;
        float x0 = rp[lane];
        float x1 = rp[lane + 32];
        float x2 = rp[lane + 64];
        float x3 = rp[lane + 96];
        float x4 = (lane < 16) ? rp[lane + 128] : -1e30f;
        float mx = fmaxf(fmaxf(fmaxf(x0, x1), fmaxf(x2, x3)), x4);
#pragma unroll
        for (int off = 16; off > 0; off >>= 1)
            mx = fmaxf(mx, __shfl_xor(mx, off, 32));
        float e0 = __expf(x0 - mx);
        float e1 = __expf(x1 - mx);
        float e2 = __expf(x2 - mx);
        float e3 = __expf(x3 - mx);
        float e4 = (lane < 16) ? __expf(x4 - mx) : 0.f;
        float s = e0 + e1 + e2 + e3 + e4;
#pragma unroll
        for (int off = 16; off > 0; off >>= 1)
            s += __shfl_xor(s, off, 32);
        float inv = 1.f / s;
        rp[lane]      = e0 * inv;
        rp[lane + 32] = e1 * inv;
        rp[lane + 64] = e2 * inv;
        rp[lane + 96] = e3 * inv;
        if (lane < 16) {
            rp[lane + 128] = e4 * inv;
            rp[LW + lane]  = 0.f;   // zero K padding cols 144..159
        }
    }
    __syncthreads();

    // ---- O = P @ V : 9 (M) x 2 (N) tiles, K = 160 (5 steps) ----
    for (int t = wave; t < 18; t += 8) {
        int mt = t >> 1;
        int nt = t & 1;
        v8f c = {};
#pragma unroll
        for (int k0 = 0; k0 < 160; k0 += 32) {
            int row = mt * 16 + (lane & 15);
            int kb  = k0 + (lane >> 4) * 8;
            const float* rp = S + row * 160;
            float4 f0 = *(const float4*)(rp + kb);
            float4 f1 = *(const float4*)(rp + kb + 4);
            float4 g0 = *(const float4*)(rp + kb + 16);
            float4 g1 = *(const float4*)(rp + kb + 20);
            v16h a;
            a[0]  = (_Float16)f0.x; a[1]  = (_Float16)f0.y;
            a[2]  = (_Float16)f0.z; a[3]  = (_Float16)f0.w;
            a[4]  = (_Float16)f1.x; a[5]  = (_Float16)f1.y;
            a[6]  = (_Float16)f1.z; a[7]  = (_Float16)f1.w;
            a[8]  = (_Float16)g0.x; a[9]  = (_Float16)g0.y;
            a[10] = (_Float16)g0.z; a[11] = (_Float16)g0.w;
            a[12] = (_Float16)g1.x; a[13] = (_Float16)g1.y;
            a[14] = (_Float16)g1.z; a[15] = (_Float16)g1.w;
            v16h bf = load_b_f16(vts + (nt * 16) * 160 + k0, 160, lane);
            c = wmma_f16(a, bf, c);
        }
        int ncol = nt * 16 + (lane & 15);
        _Float16* op = Oh + (size_t)b * LW * CC + h * DH;
#pragma unroll
        for (int r = 0; r < 8; ++r) {
            int mrow = mt * 16 + ((lane >> 4) * 8) + r;
            op[(size_t)mrow * CC + ncol] = (_Float16)c[r];
        }
    }
}

// ---------------------------------------------------------------------------
// Kernel 3: output projection  O(f16) @ Wp^T + bp -> f32
// block = 256 (8 waves), tile = 128 x 64, grid = (M/128, 512/64)
// ---------------------------------------------------------------------------
__global__ void proj_gemm_kernel(const _Float16* __restrict__ Oh,  // [M][512]
                                 const _Float16* __restrict__ Wph, // [512][512]
                                 const float* __restrict__ bp,
                                 float* __restrict__ out) {        // [M][512]
    __shared__ __align__(32) _Float16 Abuf0[128 * 32];   // 8 KB
    __shared__ __align__(32) _Float16 Abuf1[128 * 32];   // 8 KB

    const int m0   = blockIdx.x * 128;
    const int n0   = blockIdx.y * 64;
    const int tid  = threadIdx.x;
    const int wave = tid >> 5;
    const int lane = tid & 31;
    const int wm   = wave >> 1;
    const int wn   = wave & 1;

    const int arow = tid >> 1;
    const int acol = (tid & 1) * 16;
    const _Float16* osrc = Oh + (size_t)(m0 + arow) * CC + acol;
    _Float16* const d0   = Abuf0 + arow * 32 + acol;
    _Float16* const d1   = Abuf1 + arow * 32 + acol;

    const _Float16* wb0 = Wph + (size_t)(n0 + wn * 32) * CC;
    const _Float16* wb1 = Wph + (size_t)(n0 + wn * 32 + 16) * CC;
    const _Float16* a0p = Abuf0 + (wm * 32) * 32;
    const _Float16* a1p = Abuf0 + (wm * 32 + 16) * 32;
    const _Float16* a0q = Abuf1 + (wm * 32) * 32;
    const _Float16* a1q = Abuf1 + (wm * 32 + 16) * 32;

    v8f acc[2][2] = {};

    *(v8h*)(d0)     = *(const v8h*)(osrc);
    *(v8h*)(d0 + 8) = *(const v8h*)(osrc + 8);
    __syncthreads();

#pragma unroll 1
    for (int k0 = 0; k0 < CC; k0 += 64) {
        *(v8h*)(d1)     = *(const v8h*)(osrc + k0 + 32);
        *(v8h*)(d1 + 8) = *(const v8h*)(osrc + k0 + 40);
        __builtin_prefetch(osrc + k0 + 64, 0, 3);
        {
            v16h a0 = load_a_f16(a0p, 32, lane);
            v16h a1 = load_a_f16(a1p, 32, lane);
            v16h b0 = load_b_f16(wb0 + k0, CC, lane);
            v16h b1 = load_b_f16(wb1 + k0, CC, lane);
            acc[0][0] = wmma_f16(a0, b0, acc[0][0]);
            acc[0][1] = wmma_f16(a0, b1, acc[0][1]);
            acc[1][0] = wmma_f16(a1, b0, acc[1][0]);
            acc[1][1] = wmma_f16(a1, b1, acc[1][1]);
        }
        __syncthreads();
        if (k0 + 64 < CC) {
            *(v8h*)(d0)     = *(const v8h*)(osrc + k0 + 64);
            *(v8h*)(d0 + 8) = *(const v8h*)(osrc + k0 + 72);
        }
        {
            v16h a0 = load_a_f16(a0q, 32, lane);
            v16h a1 = load_a_f16(a1q, 32, lane);
            v16h b0 = load_b_f16(wb0 + k0 + 32, CC, lane);
            v16h b1 = load_b_f16(wb1 + k0 + 32, CC, lane);
            acc[0][0] = wmma_f16(a0, b0, acc[0][0]);
            acc[0][1] = wmma_f16(a0, b1, acc[0][1]);
            acc[1][0] = wmma_f16(a1, b0, acc[1][0]);
            acc[1][1] = wmma_f16(a1, b1, acc[1][1]);
        }
        __syncthreads();
    }

#pragma unroll
    for (int ms = 0; ms < 2; ++ms)
#pragma unroll
        for (int ns = 0; ns < 2; ++ns) {
            int n = n0 + wn * 32 + ns * 16 + (lane & 15);
            float bval = bp[n];
#pragma unroll
            for (int r = 0; r < 8; ++r) {
                int m = m0 + wm * 32 + ms * 16 + ((lane >> 4) * 8) + r;
                out[(size_t)m * CC + n] = acc[ms][ns][r] + bval;
            }
        }
}

// ---------------------------------------------------------------------------
extern "C" void kernel_launch(void* const* d_in, const int* in_sizes, int n_in,
                              void* d_out, int out_size, void* d_ws,
                              size_t ws_size, hipStream_t stream) {
    const float* x    = (const float*)d_in[0];
    const float* mask = (const float*)d_in[1];
    const float* Wq   = (const float*)d_in[2];
    const float* bq   = (const float*)d_in[3];
    const float* Wk   = (const float*)d_in[4];
    const float* bk   = (const float*)d_in[5];
    const float* Wv   = (const float*)d_in[6];
    const float* bv   = (const float*)d_in[7];
    const float* Wp   = (const float*)d_in[8];
    const float* bp   = (const float*)d_in[9];
    const float* lsc  = (const float*)d_in[10];
    float* out = (float*)d_out;

    char* w = (char*)d_ws;
    _Float16* Wcat = (_Float16*)(w);                          // 1,572,864 B
    _Float16* Wph  = (_Float16*)(w + 1572864);                //   524,288 B
    float*    scl  = (float*)(w + 2097152);                   //     1,024 B
    _Float16* qkvh = (_Float16*)(w + 2098176);                // 452,984,832 B
    _Float16* Ohb  = (_Float16*)(w + 2098176 + 452984832ull); // 150,994,944 B

    prep_kernel<<<dim3(1024), dim3(256), 0, stream>>>(Wq, Wk, Wv, Wp, lsc,
                                                      Wcat, Wph, scl);

    qkv_gemm_kernel<<<dim3(MM / 128, NQKV / 64), dim3(256), 0, stream>>>(
        x, Wcat, bq, bk, bv, scl, qkvh);

    attn_kernel<<<dim3(BW, HH), dim3(256), 0, stream>>>(qkvh, mask, Ohb);

    proj_gemm_kernel<<<dim3(MM / 128, CC / 64), dim3(256), 0, stream>>>(
        Ohb, Wph, bp, out);
}
